// GemmaAttention_72868415144443
// MI455X (gfx1250) — compile-verified, matrix operand order checked by reading, not run on hardware
//
#include <hip/hip_runtime.h>

// ---------------------------------------------------------------------------
// Types
// ---------------------------------------------------------------------------
typedef __attribute__((ext_vector_type(8)))  float        v8f;
typedef __attribute__((ext_vector_type(16))) __bf16       v16bf;
typedef __attribute__((ext_vector_type(4)))  unsigned int u32x4;
typedef __attribute__((ext_vector_type(2)))  unsigned int u32x2;

union Frag16 {
    v16bf          v;
    u32x4          q[2];
    unsigned short u[16];
};

static __device__ inline unsigned short f2bf(float f) {
    unsigned int u = __float_as_uint(f);
    unsigned int r = u + 0x7FFFu + ((u >> 16) & 1u);   // RNE
    return (unsigned short)(r >> 16);
}
static __device__ inline float bf2f(unsigned short s) {
    return __uint_as_float(((unsigned int)s) << 16);
}

static __device__ inline v8f wmma_bf16(v16bf a, v16bf b, v8f c) {
    return __builtin_amdgcn_wmma_f32_16x16x32_bf16(false, a, false, b,
                                                   (short)0, c, false, false);
}

// LDS byte offset of a __shared__ object (generic->LDS = low 32 bits on amdgcn)
template <typename T>
static __device__ inline unsigned lds_off(T* p) {
    return (unsigned)(unsigned long long)p;
}

// Problem constants
#define B_  2
#define T_  2048
#define C_  2048
#define H_  16
#define KV_ 4
#define D_  256

// ---------------------------------------------------------------------------
// fp32 -> bf16 convert (4 elems / thread)
// ---------------------------------------------------------------------------
__global__ __launch_bounds__(256) void f32_to_bf16_kernel(
        const float* __restrict__ in, unsigned short* __restrict__ out, long n) {
    long i = ((long)blockIdx.x * 256 + threadIdx.x) * 4;
    if (i >= n) return;
    float a = in[i + 0], b = in[i + 1], c = in[i + 2], d = in[i + 3];
    u32x2 w;
    w.x = (unsigned int)f2bf(a) | ((unsigned int)f2bf(b) << 16);
    w.y = (unsigned int)f2bf(c) | ((unsigned int)f2bf(d) << 16);
    *(u32x2*)(out + i) = w;
}

// ---------------------------------------------------------------------------
// fp32 (R x N) -> bf16 transposed (N x R), 32x32 LDS tiles
// ---------------------------------------------------------------------------
__global__ __launch_bounds__(256) void transpose_f32_to_bf16_kernel(
        const float* __restrict__ in, unsigned short* __restrict__ out,
        int R, int N) {
    __shared__ float tile[32][33];
    const int tx = threadIdx.x, ty = threadIdx.y;          // 32 x 8
    const int bx = blockIdx.x, by = blockIdx.y;
#pragma unroll
    for (int i = 0; i < 4; ++i) {
        int r = by * 32 + ty + i * 8;
        int c = bx * 32 + tx;
        tile[ty + i * 8][tx] = in[(size_t)r * N + c];
    }
    __syncthreads();
#pragma unroll
    for (int i = 0; i < 4; ++i) {
        int rO = bx * 32 + ty + i * 8;                     // N index
        int cO = by * 32 + tx;                             // R index
        out[(size_t)rO * R + cO] = f2bf(tile[tx][ty + i * 8]);
    }
}

// ---------------------------------------------------------------------------
// WMMA bf16 GEMM with async-to-LDS double-buffered staging.
//   C(MxN) = A(MxK, row-major bf16) * BT(NxK, row-major bf16)^T
//   MODE 0: C = f32 row-major   MODE 1: C = bf16 row-major
//   MODE 2: C = bf16 "V-transposed": out[(m/Tdim)*N + n][t = m%Tdim]
// Block: 256 thr = 8 waves -> 128x128 tile; wave: 32x64 (2x4 WMMA tiles).
// ---------------------------------------------------------------------------
#define LDSW 40   // padded row stride in shorts (80B)

static __device__ __forceinline__ void async_stage_tile(
        const unsigned short* __restrict__ gbase, long row0, int K, int k0,
        unsigned ldsbase, int tid) {
#pragma unroll
    for (int c2 = 0; c2 < 2; ++c2) {
        int chunk = tid * 2 + c2;          // 0..511
        int row = chunk >> 2, c16 = chunk & 3;
        unsigned voff  = (unsigned)(((size_t)(row0 + row) * K + k0 + c16 * 8) * 2);
        unsigned laddr = ldsbase + (unsigned)(row * (LDSW * 2) + c16 * 16);
        asm volatile("global_load_async_to_lds_b128 %0, %1, %2"
                     :: "v"(laddr), "v"(voff), "s"(gbase)
                     : "memory");
    }
}

template <int MODE>
__global__ __launch_bounds__(256) void gemm_bf16_kernel(
        const unsigned short* __restrict__ A,
        const unsigned short* __restrict__ BT,
        void* __restrict__ Cout,
        int M, int N, int K, int Tdim) {
    __shared__ unsigned short Abuf[2][128][LDSW];
    __shared__ unsigned short Bbuf[2][128][LDSW];

    const int tid  = threadIdx.x;
    const int wave = tid >> 5, lane = tid & 31;
    const int half = lane >> 4, lr = lane & 15;
    const int wm = wave >> 1, wn = wave & 1;
    const long bM = (long)blockIdx.y * 128;
    const long bN = (long)blockIdx.x * 128;

    const unsigned ldsA0 = lds_off(&Abuf[0][0][0]);
    const unsigned ldsA1 = lds_off(&Abuf[1][0][0]);
    const unsigned ldsB0 = lds_off(&Bbuf[0][0][0]);
    const unsigned ldsB1 = lds_off(&Bbuf[1][0][0]);

    const v8f vz = {0.f, 0.f, 0.f, 0.f, 0.f, 0.f, 0.f, 0.f};
    v8f acc[2][4];
#pragma unroll
    for (int i = 0; i < 2; ++i)
#pragma unroll
        for (int j = 0; j < 4; ++j) acc[i][j] = vz;

    const int nTiles = K / 32;
    // prologue: stage tile 0 into buffer 0
    async_stage_tile(A,  bM, K, 0, ldsA0, tid);
    async_stage_tile(BT, bN, K, 0, ldsB0, tid);

#pragma unroll 2
    for (int t = 0; t < nTiles; ++t) {
        const int buf = t & 1;
        if (t + 1 < nTiles) {
            const int k1 = (t + 1) * 32;
            async_stage_tile(A,  bM, K, k1, buf ? ldsA0 : ldsA1, tid);
            async_stage_tile(BT, bN, K, k1, buf ? ldsB0 : ldsB1, tid);
            asm volatile("s_wait_asynccnt 4" ::: "memory");  // tile t complete
        } else {
            asm volatile("s_wait_asynccnt 0" ::: "memory");
        }
        __syncthreads();   // tile t visible to all waves

        Frag16 a[2], b[4];
#pragma unroll
        for (int i = 0; i < 2; ++i) {
            const unsigned short* p = &Abuf[buf][wm * 32 + i * 16 + lr][half * 8];
            a[i].q[0] = *(const u32x4*)(p);
            a[i].q[1] = *(const u32x4*)(p + 16);
        }
#pragma unroll
        for (int j = 0; j < 4; ++j) {
            const unsigned short* p = &Bbuf[buf][wn * 64 + j * 16 + lr][half * 8];
            b[j].q[0] = *(const u32x4*)(p);
            b[j].q[1] = *(const u32x4*)(p + 16);
        }
#pragma unroll
        for (int i = 0; i < 2; ++i)
#pragma unroll
            for (int j = 0; j < 4; ++j)
                acc[i][j] = wmma_bf16(a[i].v, b[j].v, acc[i][j]);

        __syncthreads();   // done reading buf before it is overwritten
    }

#pragma unroll
    for (int i = 0; i < 2; ++i) {
#pragma unroll
        for (int j = 0; j < 4; ++j) {
            const long m0 = bM + wm * 32 + i * 16 + 8 * half;
            const long n  = bN + wn * 64 + j * 16 + lr;
#pragma unroll
            for (int p = 0; p < 8; ++p) {
                const long m = m0 + p;
                if (MODE == 0) {
                    ((float*)Cout)[m * (long)N + n] = acc[i][j][p];
                } else if (MODE == 1) {
                    ((unsigned short*)Cout)[m * (long)N + n] = f2bf(acc[i][j][p]);
                } else {
                    const long bb = m / Tdim, t = m % Tdim;
                    ((unsigned short*)Cout)[(bb * N + n) * Tdim + t] =
                        f2bf(acc[i][j][p]);
                }
            }
        }
    }
}

// ---------------------------------------------------------------------------
// RoPE, in place on bf16 [B*T][nH*D] buffer
// ---------------------------------------------------------------------------
__global__ __launch_bounds__(256) void rope_kernel(
        unsigned short* __restrict__ buf, const int* __restrict__ pos_ids,
        int nH, long total) {
    long idx = (long)blockIdx.x * 256 + threadIdx.x;
    if (idx >= total) return;
    int  d   = (int)(idx & 127);          // 0..127 (D/2)
    long r2  = idx >> 7;
    int  hh  = (int)(r2 % nH);
    long row = r2 / nH;                   // b*T + t
    float p  = (float)pos_ids[row];
    float inv = __expf(-logf(10000.f) * ((float)d / 128.f));
    float f = p * inv;
    float c = cosf(f), s = sinf(f);
    size_t i1 = (size_t)row * ((size_t)nH * D_) + (size_t)hh * D_ + d;
    size_t i2 = i1 + 128;
    float q1 = bf2f(buf[i1]);
    float q2 = bf2f(buf[i2]);
    buf[i1] = f2bf(q1 * c - q2 * s);
    buf[i2] = f2bf(q2 * c + q1 * s);
}

// ---------------------------------------------------------------------------
// Flash attention (non-causal), 4 waves / block, 32 query rows / block.
//   wave w: mt = w&1 (16-row q sub-tile), dg = w>>1 (128-wide D slice of O)
//   Q: bf16 [B*T][H*D]   K: bf16 [B*T][KV*D]   Vt: bf16 [B][KV*D][T]
//   Y: bf16 [B*T][H*D]
// K blocks (32 keys x 256) are staged into double-buffered LDS with
// global_load_async_to_lds_b128; next block's DMA overlaps current compute.
// V fragments stay direct-from-global (that loop already pipelines).
// ---------------------------------------------------------------------------
#define KROW 264   // padded K-row stride in shorts (528B = 33*16)

static __device__ __forceinline__ void async_stage_kblock(
        const unsigned short* __restrict__ Kb, int b, int kvh, int kblk,
        unsigned ldsbase, int tid) {
#pragma unroll
    for (int c8 = 0; c8 < 8; ++c8) {
        int chunk = tid * 8 + c8;          // 0..1023
        int row = chunk >> 5, c16 = chunk & 31;   // 32 chunks per 256-elem row
        unsigned voff = (unsigned)(((size_t)(b * T_ + kblk * 32 + row) * (KV_ * D_)
                                    + kvh * D_ + c16 * 8) * 2);
        unsigned laddr = ldsbase + (unsigned)(row * (KROW * 2) + c16 * 16);
        asm volatile("global_load_async_to_lds_b128 %0, %1, %2"
                     :: "v"(laddr), "v"(voff), "s"(Kb)
                     : "memory");
    }
}

__global__ __launch_bounds__(128) void attn_kernel(
        const unsigned short* __restrict__ Q,
        const unsigned short* __restrict__ Kb,
        const unsigned short* __restrict__ Vt,
        unsigned short* __restrict__ Y) {
    constexpr int LDQ = H_ * D_;    // 4096
    const int tid  = threadIdx.x;
    const int wave = tid >> 5, lane = tid & 31;
    const int half = lane >> 4, lr = lane & 15;
    const int mt = wave & 1, dg = wave >> 1;
    const int b = blockIdx.z, h = blockIdx.y, kvh = h >> 2;  // H/KV = 4
    const int q0 = blockIdx.x * 32;

    __shared__ unsigned short Kbuf[2][32][KROW];   // 2 x 16.5KB
    __shared__ float Sbuf[32][33];
    __shared__ float scalebuf[32];
    __shared__ float lbuf[32];

    const unsigned ldsK0 = lds_off(&Kbuf[0][0][0]);
    const unsigned ldsK1 = lds_off(&Kbuf[1][0][0]);

    // Preload Q fragments (16 rows x 256 = 8 K-chunks of 32)
    Frag16 qa[8];
    {
        const unsigned short* qrow =
            Q + (size_t)(b * T_ + q0 + mt * 16 + lr) * LDQ + h * D_ + half * 8;
#pragma unroll
        for (int kk = 0; kk < 8; ++kk) {
            qa[kk].q[0] = *(const u32x4*)(qrow + kk * 32);
            qa[kk].q[1] = *(const u32x4*)(qrow + kk * 32 + 16);
        }
    }

    const v8f vz = {0.f, 0.f, 0.f, 0.f, 0.f, 0.f, 0.f, 0.f};
    v8f o[8];
#pragma unroll
    for (int dt = 0; dt < 8; ++dt) o[dt] = vz;
    float m_run = -3.0e38f, l_run = 0.f;

    // prologue: stage key block 0 into buffer 0
    async_stage_kblock(Kb, b, kvh, 0, ldsK0, tid);

    for (int kb = 0; kb < T_ / 32; ++kb) {
        const int buf = kb & 1;
        const int k0 = kb * 32;
        if (kb + 1 < T_ / 32) {
            async_stage_kblock(Kb, b, kvh, kb + 1, buf ? ldsK0 : ldsK1, tid);
            asm volatile("s_wait_asynccnt 8" ::: "memory");  // block kb complete
        } else {
            asm volatile("s_wait_asynccnt 0" ::: "memory");
        }
        __syncthreads();   // key block kb visible to all waves

        // --- S tile (mt, key-subtile dg): 8 WMMAs over D, b-frags from LDS ---
        v8f s = vz;
        const unsigned short* krow = &Kbuf[buf][dg * 16 + lr][half * 8];
#pragma unroll
        for (int kk = 0; kk < 8; ++kk) {
            Frag16 kf;
            kf.q[0] = *(const u32x4*)(krow + kk * 32);
            kf.q[1] = *(const u32x4*)(krow + kk * 32 + 16);
            s = wmma_bf16(qa[kk].v, kf.v, s);
        }
#pragma unroll
        for (int p = 0; p < 8; ++p)
            Sbuf[mt * 16 + p + 8 * half][dg * 16 + lr] = s[p] * 0.0625f; // 1/sqrt(256)
        __syncthreads();

        // --- online softmax stats for row r (dup in both lane halves) ---
        const int r = mt * 16 + lr;
        float mx = -3.0e38f;
#pragma unroll 8
        for (int c = 0; c < 32; ++c) mx = fmaxf(mx, Sbuf[r][c]);
        float m_new = fmaxf(m_run, mx);
        float scl = __expf(m_run - m_new);
        float sum = 0.f;
#pragma unroll 8
        for (int c = 0; c < 32; ++c) sum += __expf(Sbuf[r][c] - m_new);
        l_run = l_run * scl + sum;
        m_run = m_new;
        if (wave < 2 && lane < 16) scalebuf[r] = scl;
        __syncthreads();

        // --- rescale O accumulators ---
        float rs[8];
#pragma unroll
        for (int p = 0; p < 8; ++p) rs[p] = scalebuf[mt * 16 + p + 8 * half];
#pragma unroll
        for (int dt = 0; dt < 8; ++dt)
#pragma unroll
            for (int p = 0; p < 8; ++p) o[dt][p] *= rs[p];

        // --- P fragment (16 x 32 keys) from LDS ---
        Frag16 pf;
#pragma unroll
        for (int e = 0; e < 16; ++e) {
            int ke = (e < 8) ? (half * 8 + e) : (16 + half * 8 + (e - 8));
            pf.u[e] = f2bf(__expf(Sbuf[r][ke] - m_run));
        }

        // --- O += P * V : 8 WMMAs over this wave's 128-wide D slice ---
        const unsigned short* vbase =
            Vt + ((size_t)(b * KV_ + kvh) * D_ + (size_t)dg * 128 + lr) * T_ +
            k0 + half * 8;
        Frag16 vf[2];
        vf[0].q[0] = *(const u32x4*)(vbase);
        vf[0].q[1] = *(const u32x4*)(vbase + 16);
#pragma unroll
        for (int dt = 0; dt < 8; ++dt) {
            if (dt < 7) {
                const unsigned short* vrow = vbase + (size_t)(dt + 1) * 16 * T_;
                vf[(dt + 1) & 1].q[0] = *(const u32x4*)(vrow);
                vf[(dt + 1) & 1].q[1] = *(const u32x4*)(vrow + 16);
            }
            o[dt] = wmma_bf16(pf.v, vf[dt & 1].v, o[dt]);
        }
        __syncthreads();   // Sbuf/scalebuf/Kbuf[buf^1] reuse fence
    }

    if (wave < 2 && lane < 16) lbuf[mt * 16 + lr] = l_run;
    __syncthreads();
    float invl[8];
#pragma unroll
    for (int p = 0; p < 8; ++p) invl[p] = 1.f / lbuf[mt * 16 + p + 8 * half];
#pragma unroll
    for (int dt = 0; dt < 8; ++dt) {
        int dglob = (dg * 8 + dt) * 16 + lr;
#pragma unroll
        for (int p = 0; p < 8; ++p) {
            size_t row = (size_t)(b * T_ + q0 + mt * 16 + p + 8 * half);
            Y[row * LDQ + h * D_ + dglob] = f2bf(o[dt][p] * invl[p]);
        }
    }
}

// ---------------------------------------------------------------------------
// Launch
// ---------------------------------------------------------------------------
extern "C" void kernel_launch(void* const* d_in, const int* in_sizes, int n_in,
                              void* d_out, int out_size, void* d_ws, size_t ws_size,
                              hipStream_t stream) {
    (void)in_sizes; (void)n_in; (void)out_size; (void)ws_size;
    const float* x   = (const float*)d_in[0];
    const int*   pos = (const int*)d_in[1];
    const float* Wq  = (const float*)d_in[2];
    const float* Wk  = (const float*)d_in[3];
    const float* Wv  = (const float*)d_in[4];
    const float* Wo  = (const float*)d_in[5];

    const long M   = (long)B_ * T_;   // 4096
    const int  Cc  = C_;              // 2048
    const int  NQ  = H_ * D_;         // 4096
    const int  NKV = KV_ * D_;        // 1024

    char* wsb = (char*)d_ws;
    size_t off = 0;
    auto take = [&](size_t bytes) -> char* {
        char* p = wsb + off;
        off += (bytes + 255) & ~(size_t)255;
        return p;
    };
    unsigned short* xb  = (unsigned short*)take((size_t)M * Cc * 2);
    unsigned short* wqT = (unsigned short*)take((size_t)NQ * Cc * 2);
    unsigned short* wkT = (unsigned short*)take((size_t)NKV * Cc * 2);
    unsigned short* wvT = (unsigned short*)take((size_t)NKV * Cc * 2);
    unsigned short* woT = (unsigned short*)take((size_t)Cc * NQ * 2);
    unsigned short* qb  = (unsigned short*)take((size_t)M * NQ * 2);
    unsigned short* kb  = (unsigned short*)take((size_t)M * NKV * 2);
    unsigned short* vtb = (unsigned short*)take((size_t)M * NKV * 2);
    unsigned short* yb  = (unsigned short*)take((size_t)M * NQ * 2);

    // 1) x -> bf16
    {
        long n = M * Cc;
        f32_to_bf16_kernel<<<(unsigned)(n / 4 / 256), 256, 0, stream>>>(x, xb, n);
    }
    // 2) weights -> bf16 transposed
    transpose_f32_to_bf16_kernel<<<dim3(NQ / 32, Cc / 32), dim3(32, 8), 0, stream>>>(Wq, wqT, Cc, NQ);
    transpose_f32_to_bf16_kernel<<<dim3(NKV / 32, Cc / 32), dim3(32, 8), 0, stream>>>(Wk, wkT, Cc, NKV);
    transpose_f32_to_bf16_kernel<<<dim3(NKV / 32, Cc / 32), dim3(32, 8), 0, stream>>>(Wv, wvT, Cc, NKV);
    transpose_f32_to_bf16_kernel<<<dim3(Cc / 32, NQ / 32), dim3(32, 8), 0, stream>>>(Wo, woT, NQ, Cc);

    // 3) projections (WMMA + async-LDS staging)
    gemm_bf16_kernel<1><<<dim3(NQ / 128, M / 128), 256, 0, stream>>>(xb, wqT, qb, (int)M, NQ, Cc, 0);
    gemm_bf16_kernel<1><<<dim3(NKV / 128, M / 128), 256, 0, stream>>>(xb, wkT, kb, (int)M, NKV, Cc, 0);
    gemm_bf16_kernel<2><<<dim3(NKV / 128, M / 128), 256, 0, stream>>>(xb, wvT, vtb, (int)M, NKV, Cc, T_);

    // 4) RoPE on q, k
    {
        long nq = M * (long)H_ * 128;
        long nk = M * (long)KV_ * 128;
        rope_kernel<<<(unsigned)((nq + 255) / 256), 256, 0, stream>>>(qb, pos, H_, nq);
        rope_kernel<<<(unsigned)((nk + 255) / 256), 256, 0, stream>>>(kb, pos, KV_, nk);
    }

    // 5) flash attention (WMMA + async-LDS K staging)
    attn_kernel<<<dim3(T_ / 32, H_, B_), 128, 0, stream>>>(qb, kb, vtb, yb);

    // 6) output projection (WMMA, f32 out)
    gemm_bf16_kernel<0><<<dim3(Cc / 128, M / 128), 256, 0, stream>>>(yb, woT, d_out, (int)M, Cc, NQ, 0);
}
